// BaseGR_51788715655933
// MI455X (gfx1250) — matrix operand order repaired and checked by reading, not compiled, not run on hardware
//
#include <hip/hip_runtime.h>
#include <hip/hip_bf16.h>

// Problem constants (match reference setup_inputs)
constexpr int HD   = 128;
constexpr int NGc  = 5000;
constexpr int NUc  = 100000;
constexpr int NIc  = 20000;
constexpr long EUG = 300000;
constexpr long EUI = 600000;
constexpr long EGI = 200000;

typedef __attribute__((ext_vector_type(16))) __bf16 v16bf;
typedef __attribute__((ext_vector_type(8)))  float  v8f;
typedef unsigned short ushort_t;

__device__ __forceinline__ ushort_t f2bf(float f) {
  union { float f; unsigned u; } v; v.f = f;
  unsigned r = v.u + 0x7FFFu + ((v.u >> 16) & 1u);  // round-to-nearest-even
  return (ushort_t)(r >> 16);
}
__device__ __forceinline__ float bf2f(ushort_t h) {
  union { float f; unsigned u; } v; v.u = ((unsigned)h) << 16;
  return v.f;
}

union AFrag {
  v16bf v;
  ushort_t u[16];
  uint4 q[2];
};

// ---------------------------------------------------------------- utilities
__global__ void zero_f32(float* __restrict__ p, long n) {
  long i = (long)blockIdx.x * blockDim.x + threadIdx.x;
  long stride = (long)gridDim.x * blockDim.x;
  for (; i < n; i += stride) p[i] = 0.0f;
}

// One-shot f32 -> bf16 conversion (features / weights / pred_W)
__global__ void f32_to_bf16(const float* __restrict__ in, ushort_t* __restrict__ out, long n) {
  long i = (long)blockIdx.x * blockDim.x + threadIdx.x;
  long stride = (long)gridDim.x * blockDim.x;
  for (; i < n; i += stride) out[i] = f2bf(in[i]);
}

// One edge per group of 32 lanes: gather bf16 x[src] row, atomic-add f32 into
// accum[dst] row, count in-degree.
__global__ __launch_bounds__(256) void scatter_add(
    const ushort_t* __restrict__ x, const int* __restrict__ src,
    const int* __restrict__ dst, float* __restrict__ accum,
    float* __restrict__ cnt, int E)
{
  long t = (long)blockIdx.x * 256 + threadIdx.x;
  long e = t >> 5;
  int lane = (int)(t & 31);
  if (e >= E) return;
  int s = src[e];
  int d = dst[e];
  uint2 xv = *(const uint2*)(x + (size_t)s * HD + lane * 4);  // 4 bf16
  float* ap = accum + (size_t)d * HD + lane * 4;
  atomicAdd(ap + 0, bf2f((ushort_t)(xv.x & 0xFFFFu)));
  atomicAdd(ap + 1, bf2f((ushort_t)(xv.x >> 16)));
  atomicAdd(ap + 2, bf2f((ushort_t)(xv.y & 0xFFFFu)));
  atomicAdd(ap + 3, bf2f((ushort_t)(xv.y >> 16)));
  if (lane == 0) atomicAdd(cnt + d, 1.0f);
}

// mean = accum / max(cnt,1), emitted as bf16 (GEMM A operand format)
__global__ void seg_div_bf16(const float* __restrict__ accum, const float* __restrict__ cnt,
                             ushort_t* __restrict__ out, int n) {
  long i = (long)blockIdx.x * blockDim.x + threadIdx.x;
  long total = (long)n * HD;
  if (i >= total) return;
  float c = cnt[i >> 7];
  out[i] = f2bf(accum[i] / fmaxf(c, 1.0f));
}

// ---------------------------------------------------------------- WMMA GEMM
// D[M x N] = relu?( sum_t A_t[M x 128]bf16 @ B_t[128 x N]bf16 + bias0 + bias1 )
// Block = 256 threads (8 wave32s); tile 128 rows x 128 cols; wave w owns rows
// [mb0+16w, +16) x 128 cols (8 accum tiles). B tile staged transposed in LDS.
// A fragments load straight from bf16 global memory in the ISA register layout
// (16-bit A 16x32: lanes 0-15 kbase=0, lanes 16-31 kbase=8; split-K halves).
constexpr int LDL = 136;  // padded LDS row stride (ushorts); 272 B, 16B-aligned

__global__ __launch_bounds__(256) void gemm4(
    const ushort_t* __restrict__ A0, const ushort_t* __restrict__ B0,
    const ushort_t* __restrict__ A1, const ushort_t* __restrict__ B1,
    const ushort_t* __restrict__ A2, const ushort_t* __restrict__ B2,
    const ushort_t* __restrict__ A3, const ushort_t* __restrict__ B3,
    const float* __restrict__ bias0, const float* __restrict__ bias1,
    float* __restrict__ outF, ushort_t* __restrict__ outB,
    int M, int N, int relu)
{
  __shared__ __align__(16) ushort_t lB[128 * LDL];

  const int tid    = threadIdx.x;
  const int wave   = tid >> 5;      // wave32
  const int lane   = tid & 31;
  const int nlo    = lane & 15;
  const int lanehi = lane >> 4;     // 0 | 1
  const int kbase  = lanehi * 8;    // A-frag K sub-offset
  const int koff   = lanehi * 8;    // B-frag K sub-offset
  const int mb0    = blockIdx.x * 128;
  const int nb0    = blockIdx.y * 128;
  const int mwave  = mb0 + wave * 16;
  const int mrow   = mwave + nlo;   // A row this lane feeds

  v8f acc[8];
  v8f zf = {};
  for (int nt = 0; nt < 8; ++nt) acc[nt] = zf;

  for (int t = 0; t < 4; ++t) {
    const ushort_t* A; const ushort_t* B;
    if      (t == 0) { A = A0; B = B0; }
    else if (t == 1) { A = A1; B = B1; }
    else if (t == 2) { A = A2; B = B2; }
    else             { A = A3; B = B3; }
    if (A == nullptr) continue;  // uniform (SGPR) branch

    // Stage B tile transposed into LDS: lB[n][k] (pure bf16 copy, uint reads)
    for (int idx = tid; idx < 128 * 64; idx += 256) {
      int k  = idx >> 6;
      int n2 = (idx & 63) * 2;
      int gn = nb0 + n2;
      unsigned v = 0;
      if (gn + 1 < N)     v = *(const unsigned*)(B + (size_t)k * N + gn);
      else if (gn < N)    v = (unsigned)B[(size_t)k * N + gn];
      lB[n2 * LDL + k]       = (ushort_t)(v & 0xFFFFu);
      lB[(n2 + 1) * LDL + k] = (ushort_t)(v >> 16);
    }
    __syncthreads();

    const ushort_t* arow = A + (size_t)mrow * HD;
    const bool mvalid = (mrow < M);
    for (int kt = 0; kt < 4; ++kt) {       // K = 128 in 4 steps of 32
      const int k0 = kt * 32 + kbase;
      AFrag af;
      if (mvalid) {                        // two 16B runs, no conversion
        af.q[0] = *(const uint4*)(arow + k0);
        af.q[1] = *(const uint4*)(arow + k0 + 16);
      } else {
        af.q[0] = make_uint4(0, 0, 0, 0);
        af.q[1] = make_uint4(0, 0, 0, 0);
      }
      // double-buffered B fragments: overlap ds_load with WMMA
      const ushort_t* bbase = &lB[nlo * LDL + kt * 32 + koff];
      AFrag bf0, bf1;
      bf0.q[0] = *(const uint4*)(bbase);
      bf0.q[1] = *(const uint4*)(bbase + 16);
#pragma unroll
      for (int nt = 0; nt < 8; ++nt) {
        AFrag& cur = (nt & 1) ? bf1 : bf0;
        AFrag& nxt = (nt & 1) ? bf0 : bf1;
        if (nt < 7) {
          const ushort_t* bp = bbase + (size_t)(nt + 1) * 16 * LDL;
          nxt.q[0] = *(const uint4*)(bp);
          nxt.q[1] = *(const uint4*)(bp + 16);
        }
        acc[nt] = __builtin_amdgcn_wmma_f32_16x16x32_bf16(
            false, af.v, false, cur.v, (short)0, acc[nt], false, false);
      }
    }
    __syncthreads();
  }

  // Epilogue: bias + optional relu; f32 (final output) or bf16 (activations).
  for (int nt = 0; nt < 8; ++nt) {
    int col = nb0 + nt * 16 + nlo;
    if (col >= N) continue;
    float b = 0.0f;
    if (bias0) b += bias0[col];
    if (bias1) b += bias1[col];
    for (int r = 0; r < 8; ++r) {
      int row = mwave + r + lanehi * 8;   // C/D layout: VGPR r -> M = r + 8*hi
      if (row < M) {
        float v = acc[nt][r] + b;
        if (relu) v = fmaxf(v, 0.0f);
        if (outF) outF[(size_t)row * N + col] = v;
        else      outB[(size_t)row * N + col] = f2bf(v);
      }
    }
  }
}

// ---------------------------------------------------------------- driver
extern "C" void kernel_launch(void* const* d_in, const int* in_sizes, int n_in,
                              void* d_out, int out_size, void* d_ws, size_t ws_size,
                              hipStream_t stream)
{
  (void)in_sizes; (void)n_in; (void)out_size; (void)ws_size;
  const float* emb_user = (const float*)d_in[4];
  const float* emb_item = (const float*)d_in[5];
  const float* W1l = (const float*)d_in[6];
  const float* W1r = (const float*)d_in[7];
  const float* b1  = (const float*)d_in[8];
  const float* W2l = (const float*)d_in[9];
  const float* W2r = (const float*)d_in[10];
  const float* b2  = (const float*)d_in[11];
  const float* pred_W = (const float*)d_in[12];
  const float* pred_b = (const float*)d_in[13];
  const int* ug_src = (const int*)d_in[14];
  const int* ug_dst = (const int*)d_in[15];
  const int* ui_src = (const int*)d_in[16];
  const int* ui_dst = (const int*)d_in[17];
  const int* gi_src = (const int*)d_in[18];
  const int* gi_dst = (const int*)d_in[19];

  // ---- workspace carve-up: f32 region first (16B multiple), then bf16.
  char* base = (char*)d_ws;
  auto allocF = [&](size_t n) { float* p = (float*)base; base += n * sizeof(float); return p; };
  auto allocH = [&](size_t n) { ushort_t* p = (ushort_t*)base; base += n * sizeof(ushort_t); return p; };

  float* mean_ug_f = allocF((size_t)NGc * HD);
  float* mean_ig_f = allocF((size_t)NGc * HD);
  float* mean_iu_f = allocF((size_t)NUc * HD);
  float* mean_ui_f = allocF((size_t)NIc * HD);
  float* cnt_gu = allocF(NGc);
  float* cnt_gi = allocF(NGc);
  float* cnt_u  = allocF(NUc);
  float* cnt_i  = allocF(NIc + 8);  // keep 16B alignment for bf16 region

  ushort_t* bemb_u  = allocH((size_t)NUc * HD);
  ushort_t* bemb_i  = allocH((size_t)NIc * HD);
  ushort_t* bW1l    = allocH((size_t)6 * HD * HD);
  ushort_t* bW1r    = allocH((size_t)6 * HD * HD);
  ushort_t* bW2l    = allocH((size_t)6 * HD * HD);
  ushort_t* bW2r    = allocH((size_t)6 * HD * HD);
  ushort_t* bpredW  = allocH((size_t)HD * NIc);
  ushort_t* bmean_ug = allocH((size_t)NGc * HD);
  ushort_t* bmean_ig = allocH((size_t)NGc * HD);
  ushort_t* bmean_iu = allocH((size_t)NUc * HD);
  ushort_t* bmean_ui = allocH((size_t)NIc * HD);
  ushort_t* bhg1 = allocH((size_t)NGc * HD);
  ushort_t* bhu1 = allocH((size_t)NUc * HD);
  ushort_t* bhi1 = allocH((size_t)NIc * HD);
  ushort_t* brep = allocH((size_t)NGc * HD);

  dim3 blk(256);

  // ---- one-shot f32 -> bf16 conversions (inputs are never mutated)
  f32_to_bf16<<<2048, 256, 0, stream>>>(emb_user, bemb_u, (long)NUc * HD);
  f32_to_bf16<<<512, 256, 0, stream>>>(emb_item, bemb_i, (long)NIc * HD);
  f32_to_bf16<<<96, 256, 0, stream>>>(W1l, bW1l, (long)6 * HD * HD);
  f32_to_bf16<<<96, 256, 0, stream>>>(W1r, bW1r, (long)6 * HD * HD);
  f32_to_bf16<<<96, 256, 0, stream>>>(W2l, bW2l, (long)6 * HD * HD);
  f32_to_bf16<<<96, 256, 0, stream>>>(W2r, bW2r, (long)6 * HD * HD);
  f32_to_bf16<<<512, 256, 0, stream>>>(pred_W, bpredW, (long)HD * NIc);

  // ---- layer 1 seg-means (hg == 0, so g->u and g->i means are skipped)
  long zcount = (long)((float*)bemb_u - mean_ug_f);  // contiguous f32 region
  zero_f32<<<2048, 256, 0, stream>>>(mean_ug_f, zcount);

  auto sgrid = [](long E) { return dim3((unsigned)((E * 32 + 255) / 256)); };
  scatter_add<<<sgrid(EUG), blk, 0, stream>>>(bemb_u, ug_src, ug_dst, mean_ug_f, cnt_gu, (int)EUG);
  scatter_add<<<sgrid(EGI), blk, 0, stream>>>(bemb_i, gi_dst, gi_src, mean_ig_f, cnt_gi, (int)EGI);
  scatter_add<<<sgrid(EUI), blk, 0, stream>>>(bemb_i, ui_dst, ui_src, mean_iu_f, cnt_u,  (int)EUI);
  scatter_add<<<sgrid(EUI), blk, 0, stream>>>(bemb_u, ui_src, ui_dst, mean_ui_f, cnt_i,  (int)EUI);

  auto dgrid = [](long n) { return dim3((unsigned)((n * HD + 255) / 256)); };
  seg_div_bf16<<<dgrid(NGc), blk, 0, stream>>>(mean_ug_f, cnt_gu, bmean_ug, NGc);
  seg_div_bf16<<<dgrid(NGc), blk, 0, stream>>>(mean_ig_f, cnt_gi, bmean_ig, NGc);
  seg_div_bf16<<<dgrid(NUc), blk, 0, stream>>>(mean_iu_f, cnt_u,  bmean_iu, NUc);
  seg_div_bf16<<<dgrid(NIc), blk, 0, stream>>>(mean_ui_f, cnt_i,  bmean_ui, NIc);

  // ---- layer 1 fused GEMMs (+bias, relu) -> bf16 activations
  {
    dim3 g((NGc + 127) / 128, 1);
    gemm4<<<g, blk, 0, stream>>>(bmean_ug, bW1l + 0 * HD * HD, bmean_ig, bW1l + 5 * HD * HD,
                                 nullptr, nullptr, nullptr, nullptr,
                                 b1 + 0 * HD, b1 + 5 * HD, nullptr, bhg1, NGc, HD, 1);
  }
  {
    dim3 g((NUc + 127) / 128, 1);
    gemm4<<<g, blk, 0, stream>>>(bemb_u, bW1r + 1 * HD * HD, bemb_u, bW1r + 3 * HD * HD,
                                 bmean_iu, bW1l + 3 * HD * HD, nullptr, nullptr,
                                 b1 + 1 * HD, b1 + 3 * HD, nullptr, bhu1, NUc, HD, 1);
  }
  {
    dim3 g((NIc + 127) / 128, 1);
    gemm4<<<g, blk, 0, stream>>>(bmean_ui, bW1l + 2 * HD * HD, bemb_i, bW1r + 2 * HD * HD,
                                 bemb_i, bW1r + 4 * HD * HD, nullptr, nullptr,
                                 b1 + 2 * HD, b1 + 4 * HD, nullptr, bhi1, NIc, HD, 1);
  }

  // ---- layer 2: only og (group output) feeds the final prediction
  zero_f32<<<128, 256, 0, stream>>>(mean_ug_f, (long)NGc * HD);
  zero_f32<<<128, 256, 0, stream>>>(mean_ig_f, (long)NGc * HD);
  zero_f32<<<8, 256, 0, stream>>>(cnt_gu, NGc);
  zero_f32<<<8, 256, 0, stream>>>(cnt_gi, NGc);

  scatter_add<<<sgrid(EUG), blk, 0, stream>>>(bhu1, ug_src, ug_dst, mean_ug_f, cnt_gu, (int)EUG);
  scatter_add<<<sgrid(EGI), blk, 0, stream>>>(bhi1, gi_dst, gi_src, mean_ig_f, cnt_gi, (int)EGI);
  seg_div_bf16<<<dgrid(NGc), blk, 0, stream>>>(mean_ug_f, cnt_gu, bmean_ug, NGc);
  seg_div_bf16<<<dgrid(NGc), blk, 0, stream>>>(mean_ig_f, cnt_gi, bmean_ig, NGc);

  {
    dim3 g((NGc + 127) / 128, 1);
    gemm4<<<g, blk, 0, stream>>>(bmean_ug, bW2l + 0 * HD * HD, bmean_ig, bW2l + 5 * HD * HD,
                                 bhg1, bW2r + 0 * HD * HD, bhg1, bW2r + 5 * HD * HD,
                                 b2 + 0 * HD, b2 + 5 * HD, nullptr, brep, NGc, HD, 1);
  }

  // ---- final prediction: out[NG x NI] = rep @ pred_W + pred_b  (f32 out)
  {
    dim3 g((NGc + 127) / 128, (NIc + 127) / 128);
    gemm4<<<g, blk, 0, stream>>>(brep, bpredW, nullptr, nullptr,
                                 nullptr, nullptr, nullptr, nullptr,
                                 pred_b, nullptr, (float*)d_out, nullptr, NGc, NIc, 0);
  }
}
